// GConv_86268713107900
// MI455X (gfx1250) — matrix operand order repaired and checked by reading, not compiled, not run on hardware
//
#include <hip/hip_runtime.h>

#define DH 1024

typedef __attribute__((ext_vector_type(16))) __bf16 v16bf;
typedef __attribute__((ext_vector_type(8)))  __bf16 v8bf;
typedef __attribute__((ext_vector_type(8)))  float  v8f;

__device__ __forceinline__ unsigned short f2bf(float f) {
    unsigned x = __float_as_uint(f);
    x += 0x7FFFu + ((x >> 16) & 1u);   // round-to-nearest-even
    return (unsigned short)(x >> 16);
}

__device__ __forceinline__ void atomAddF(float* p, float v) {
    __hip_atomic_fetch_add(p, v, __ATOMIC_RELAXED, __HIP_MEMORY_SCOPE_AGENT);
}

// ---------------------------------------------------------------- copy (f32)
__global__ void copy_f32(float* __restrict__ dst, const float* __restrict__ src, int n4) {
    int i = blockIdx.x * blockDim.x + threadIdx.x;
    int stride = gridDim.x * blockDim.x;
    const float4* s = (const float4*)src;
    float4* d = (float4*)dst;
    for (; i < n4; i += stride) d[i] = s[i];
}

// ------------------------------------------------- edge scatter-add (1 wave/edge)
__global__ void scatter_add(const float* __restrict__ z, const int* __restrict__ srcI,
                            const int* __restrict__ dstI, float* __restrict__ agg,
                            int E, int D4) {
    int lane = threadIdx.x & 31;
    int wave = (blockIdx.x * blockDim.x + threadIdx.x) >> 5;
    int nwaves = (gridDim.x * blockDim.x) >> 5;
    for (int e = wave; e < E; e += nwaves) {
        int s = srcI[e];
        int d = dstI[e];
        const float4* zp = (const float4*)(z + (size_t)s * (size_t)(D4 * 4));
        float* ap = agg + (size_t)d * (size_t)(D4 * 4);
        for (int c = lane; c < D4; c += 32) {
            float4 v = zp[c];
            atomAddF(ap + 4 * c + 0, v.x);
            atomAddF(ap + 4 * c + 1, v.y);
            atomAddF(ap + 4 * c + 2, v.z);
            atomAddF(ap + 4 * c + 3, v.w);
        }
    }
}

// ---------------------------------------------------------------- f32 -> bf16
__global__ void cvt_bf16(const float* __restrict__ src, unsigned short* __restrict__ dst, int n4) {
    int i = blockIdx.x * blockDim.x + threadIdx.x;
    int stride = gridDim.x * blockDim.x;
    for (; i < n4; i += stride) {
        float4 v = ((const float4*)src)[i];
        ushort4 o;
        o.x = f2bf(v.x); o.y = f2bf(v.y); o.z = f2bf(v.z); o.w = f2bf(v.w);
        ((ushort4*)dst)[i] = o;
    }
}

// -------------------- pack weight W[K][1024] (f32 row-major) into WMMA-B lane layout
// B tile (kt,nt): lane n in 0..15 = column nt*16+n holding k = kt*32 + 0..7,16..23,
// lane 16..31 = same column holding k = kt*32 + 8..15,24..31; 16 bf16 per lane.
__global__ void pack_w(const float* __restrict__ W, unsigned short* __restrict__ Wp, int K) {
    int t = blockIdx.x * blockDim.x + threadIdx.x;
    int KT = K >> 5;
    int total = 64 * KT * 32;
    if (t >= total) return;
    int lane = t & 31;
    int tile = t >> 5;          // tile = nt*KT + kt
    int kt = tile % KT;
    int nt = tile / KT;
    int n = nt * 16 + (lane & 15);
    int kbase = kt * 32 + ((lane >> 4) << 3);
    unsigned short* out = Wp + (size_t)t * 16;
#pragma unroll
    for (int j = 0; j < 16; ++j)
        out[j] = f2bf(W[(size_t)(kbase + j + (j >= 8 ? 8 : 0)) * DH + n]);
}

// ------------------------------------------------------------------- WMMA GEMM
// out[N x 1024] = PReLU(A_bf16[N x K] @ W + bias, slope); 128x64 block tile,
// 8 waves, each wave a 32x32 tile = 4 x v_wmma_f32_16x16x32_bf16 per K-step.
template <int STORE_BF16>
__global__ __launch_bounds__(256) void gemm_wmma(
        const unsigned short* __restrict__ A, const unsigned short* __restrict__ Bp,
        const float* __restrict__ bias, const float* __restrict__ slope_p,
        void* __restrict__ out, int Nrows, int K) {
    const int lane = threadIdx.x & 31;
    const int wave = threadIdx.x >> 5;
    const int wm = wave >> 1;
    const int wn = wave & 1;
    const int m = lane & 15;
    const int hi = lane >> 4;
    const int rowBase = blockIdx.y * 128 + wm * 32;
    const int colBase = blockIdx.x * 64 + wn * 32;
    const int KT = K >> 5;

    // A layout: lane m holds row (rowBase+tile*16+m); hi-lanes take K offset +8.
    // Per 16x32 tile that is two contiguous 16B row segments per lane.
    int rA0 = rowBase + m;       if (rA0 >= Nrows) rA0 = Nrows - 1;
    int rA1 = rowBase + 16 + m;  if (rA1 >= Nrows) rA1 = Nrows - 1;
    const unsigned short* a0p = A + (size_t)rA0 * K + hi * 8;
    const unsigned short* a1p = A + (size_t)rA1 * K + hi * 8;

    const int nt0 = colBase >> 4;
    const unsigned short* b0p = Bp + ((size_t)(nt0 * KT) * 32 + lane) * 16;
    const unsigned short* b1p = Bp + ((size_t)((nt0 + 1) * KT) * 32 + lane) * 16;

    v8f acc00{}, acc01{}, acc10{}, acc11{};

    for (int kt = 0; kt < KT; ++kt) {
        const int ko = kt * 32;
        v8bf a0lo = *(const v8bf*)(a0p + ko);
        v8bf a0hi = *(const v8bf*)(a0p + ko + 16);
        v8bf a1lo = *(const v8bf*)(a1p + ko);
        v8bf a1hi = *(const v8bf*)(a1p + ko + 16);
        v16bf A0 = __builtin_shufflevector(a0lo, a0hi, 0, 1, 2, 3, 4, 5, 6, 7,
                                           8, 9, 10, 11, 12, 13, 14, 15);
        v16bf A1 = __builtin_shufflevector(a1lo, a1hi, 0, 1, 2, 3, 4, 5, 6, 7,
                                           8, 9, 10, 11, 12, 13, 14, 15);
        v16bf B0 = *(const v16bf*)(b0p + (size_t)kt * 512);
        v16bf B1 = *(const v16bf*)(b1p + (size_t)kt * 512);

        acc00 = __builtin_amdgcn_wmma_f32_16x16x32_bf16(false, A0, false, B0,
                                                        (short)0, acc00, false, false);
        acc01 = __builtin_amdgcn_wmma_f32_16x16x32_bf16(false, A0, false, B1,
                                                        (short)0, acc01, false, false);
        acc10 = __builtin_amdgcn_wmma_f32_16x16x32_bf16(false, A1, false, B0,
                                                        (short)0, acc10, false, false);
        acc11 = __builtin_amdgcn_wmma_f32_16x16x32_bf16(false, A1, false, B1,
                                                        (short)0, acc11, false, false);
    }

    const float s = slope_p[0];
    const int c0 = colBase + m;
    const int c1 = colBase + 16 + m;
    const float bv0 = bias[c0];
    const float bv1 = bias[c1];

    auto store = [&](v8f acc, int mTile, int col, float bv) {
#pragma unroll
        for (int r = 0; r < 8; ++r) {
            int row = rowBase + mTile * 16 + hi * 8 + r;   // C layout: VGPR r, hi-lanes +8
            if (row < Nrows) {
                float v = acc[r] + bv;
                v = (v >= 0.0f) ? v : s * v;
                if (STORE_BF16)
                    ((unsigned short*)out)[(size_t)row * DH + col] = f2bf(v);
                else
                    ((float*)out)[(size_t)row * DH + col] = v;
            }
        }
    };
    store(acc00, 0, c0, bv0);
    store(acc01, 0, c1, bv1);
    store(acc10, 1, c0, bv0);
    store(acc11, 1, c1, bv1);
}

// ---------------------------------------------------------------- BN pieces
__global__ void zero_f32(float* __restrict__ p, int n) {
    int i = blockIdx.x * blockDim.x + threadIdx.x;
    if (i < n) p[i] = 0.0f;
}

__global__ void bn_stats(const float* __restrict__ h, float* __restrict__ stats, int Nrows) {
    int t = threadIdx.x;  // 256 threads; each owns cols t, t+256, t+512, t+768
    float s0 = 0, s1 = 0, s2 = 0, s3 = 0;
    float q0 = 0, q1 = 0, q2 = 0, q3 = 0;
    for (int r = blockIdx.x; r < Nrows; r += gridDim.x) {
        const float* hp = h + (size_t)r * DH;
        float v0 = hp[t], v1 = hp[t + 256], v2 = hp[t + 512], v3 = hp[t + 768];
        s0 += v0; q0 += v0 * v0;
        s1 += v1; q1 += v1 * v1;
        s2 += v2; q2 += v2 * v2;
        s3 += v3; q3 += v3 * v3;
    }
    atomAddF(&stats[t],         s0); atomAddF(&stats[t + 256],  s1);
    atomAddF(&stats[t + 512],   s2); atomAddF(&stats[t + 768],  s3);
    atomAddF(&stats[1024 + t],       q0); atomAddF(&stats[1024 + t + 256], q1);
    atomAddF(&stats[1024 + t + 512], q2); atomAddF(&stats[1024 + t + 768], q3);
}

__global__ void bn_finalize(const float* __restrict__ stats, const float* __restrict__ g,
                            const float* __restrict__ be, float* __restrict__ ss, int Nrows) {
    int c = blockIdx.x * blockDim.x + threadIdx.x;
    if (c < DH) {
        float inv_n = 1.0f / (float)Nrows;
        float mu = stats[c] * inv_n;
        float var = stats[1024 + c] * inv_n - mu * mu;
        if (var < 0.0f) var = 0.0f;
        float sc = g[c] * rsqrtf(var + 1e-5f);
        ss[c] = sc;
        ss[1024 + c] = be[c] - mu * sc;
    }
}

__global__ void bn_apply(const float* __restrict__ h, const float* __restrict__ ss,
                         float* __restrict__ out, int n4) {
    int i = blockIdx.x * blockDim.x + threadIdx.x;
    int stride = gridDim.x * blockDim.x;
    for (; i < n4; i += stride) {
        float4 v = ((const float4*)h)[i];
        int c = (i * 4) & (DH - 1);
        float4 sc = *(const float4*)(ss + c);
        float4 sh = *(const float4*)(ss + 1024 + c);
        v.x = v.x * sc.x + sh.x;
        v.y = v.y * sc.y + sh.y;
        v.z = v.z * sc.z + sh.z;
        v.w = v.w * sc.w + sh.w;
        ((float4*)out)[i] = v;
    }
}

// ------------------------------------------------------------------- launcher
extern "C" void kernel_launch(void* const* d_in, const int* in_sizes, int n_in,
                              void* d_out, int out_size, void* d_ws, size_t ws_size,
                              hipStream_t stream) {
    const float* x    = (const float*)d_in[0];
    const int*   ei   = (const int*)d_in[1];
    const int E = in_sizes[1] / 2;
    const int Nn = in_sizes[0] / 768;
    const int* srcI = ei;
    const int* dstI = ei + E;

    const float* W1_0 = (const float*)d_in[3];
    const float* b1_0 = (const float*)d_in[4];
    const float* a1_0 = (const float*)d_in[5];
    const float* W2_0 = (const float*)d_in[6];
    const float* b2_0 = (const float*)d_in[7];
    const float* g_0  = (const float*)d_in[8];
    const float* be_0 = (const float*)d_in[9];
    const float* W1_1 = (const float*)d_in[10];
    const float* b1_1 = (const float*)d_in[11];
    const float* a1_1 = (const float*)d_in[12];
    const float* W2_1 = (const float*)d_in[13];
    const float* b2_1 = (const float*)d_in[14];
    const float* g_1  = (const float*)d_in[15];
    const float* be_1 = (const float*)d_in[16];
    const float* actA = (const float*)d_in[17];

    // workspace carve-up (256B aligned)
    char* ws = (char*)d_ws;
    size_t off = 0;
    auto walloc = [&](size_t bytes) {
        void* p = ws + off;
        off += (bytes + 255) & ~(size_t)255;
        return p;
    };
    float*          bufA  = (float*)walloc((size_t)Nn * DH * 4);
    float*          bufB  = (float*)walloc((size_t)Nn * DH * 4);
    unsigned short* Hb    = (unsigned short*)walloc((size_t)Nn * DH * 2);
    unsigned short* H1b   = (unsigned short*)walloc((size_t)Nn * DH * 2);
    unsigned short* Wp    = (unsigned short*)walloc((size_t)DH * DH * 2);
    float*          stats = (float*)walloc(2048 * 4);
    float*          ss    = (float*)walloc(2048 * 4);

    const int TB = 256;
    dim3 gemmGrid(DH / 64, (Nn + 127) / 128);

    auto run_layer = [&](const float* zin, int K,
                         const float* W1, const float* b1, const float* a1,
                         const float* W2, const float* b2,
                         const float* g, const float* be, float* zout) {
        const int n4in = Nn * K / 4;
        const int n4h  = Nn * DH / 4;
        // agg = z ; agg[dst] += z[src]
        copy_f32<<<(n4in + TB - 1) / TB, TB, 0, stream>>>(bufB, zin, n4in);
        scatter_add<<<16384, TB, 0, stream>>>(zin, srcI, dstI, bufB, E, K / 4);
        // bf16 activations
        cvt_bf16<<<(n4in + TB - 1) / TB, TB, 0, stream>>>(bufB, Hb, n4in);
        // GEMM1: H1b = PReLU(Hb @ W1 + b1, a1) stored bf16
        int pk1 = 64 * (K >> 5) * 32;
        pack_w<<<(pk1 + TB - 1) / TB, TB, 0, stream>>>(W1, Wp, K);
        gemm_wmma<1><<<gemmGrid, TB, 0, stream>>>(Hb, Wp, b1, a1, (void*)H1b, Nn, K);
        // GEMM2: bufB = PReLU(H1b @ W2 + b2, act_a) stored f32
        int pk2 = 64 * (DH >> 5) * 32;
        pack_w<<<(pk2 + TB - 1) / TB, TB, 0, stream>>>(W2, Wp, DH);
        gemm_wmma<0><<<gemmGrid, TB, 0, stream>>>(H1b, Wp, b2, actA, (void*)bufB, Nn, DH);
        // BatchNorm
        zero_f32<<<(2048 + TB - 1) / TB, TB, 0, stream>>>(stats, 2048);
        bn_stats<<<2048, TB, 0, stream>>>(bufB, stats, Nn);
        bn_finalize<<<(DH + TB - 1) / TB, TB, 0, stream>>>(stats, g, be, ss, Nn);
        bn_apply<<<(n4h + TB - 1) / TB, TB, 0, stream>>>(bufB, ss, zout, n4h);
    };

    // layer 0: x (K=768) -> bufA
    run_layer(x, 768, W1_0, b1_0, a1_0, W2_0, b2_0, g_0, be_0, bufA);
    // layer 1: bufA (K=1024) -> d_out
    run_layer(bufA, DH, W1_1, b1_1, a1_1, W2_1, b2_1, g_1, be_1, (float*)d_out);
}
